// TransformerLayer_40845138985516
// MI455X (gfx1250) — matrix operand order repaired and checked by reading, not compiled, not run on hardware
//
#include <hip/hip_runtime.h>

// ---------- problem constants ----------
#define E_   512
#define H_   8
#define D_   64
#define B_   64
#define LMAX_ 384
#define T_   16384
#define F_   1536   // 3*E

// ---------- workspace layout (bytes) ----------
#define WS_XN    ((size_t)0)                       // T*E bf16      = 16,777,216
#define WS_QKV   ((size_t)16777216)                // T*F bf16      = 50,331,648
#define WS_VT    ((size_t)67108864)                // B*H*D*LMAX bf16 = 25,165,824
#define WS_OUT16 ((size_t)92274688)                // B*LMAX*E bf16 = 25,165,824
#define WS_WIN   ((size_t)117440512)               // F*E bf16      = 1,572,864
#define WS_WOUT  ((size_t)119013376)               // E*E bf16      = 524,288
#define WS_OFFS  ((size_t)119537664)               // B int32

typedef __attribute__((ext_vector_type(16))) __bf16 v16bf;
typedef __attribute__((ext_vector_type(8)))  float  v8f;
typedef __attribute__((ext_vector_type(4)))  unsigned int u32x4;

#if defined(__has_builtin)
#if __has_builtin(__builtin_amdgcn_global_load_async_to_lds_b128)
#define HAVE_ASYNC_LDS 1
#endif
#endif

#ifdef HAVE_ASYNC_LDS
// builtin signature (from hipcc diagnostic): (v4i addrspace(1)*, v4i addrspace(3)*, imm int, imm int)
typedef int v4i __attribute__((vector_size(16)));
typedef __attribute__((address_space(1))) v4i gv4i;
typedef __attribute__((address_space(3))) v4i lv4i;
#endif

// float -> bf16 with round-to-nearest-even (bit trick; inputs finite)
static __device__ __forceinline__ __bf16 tobf(float f) {
    unsigned u = __builtin_bit_cast(unsigned, f);
    u += 0x7FFFu + ((u >> 16) & 1u);
    return __builtin_bit_cast(__bf16, (unsigned short)(u >> 16));
}
static __device__ __forceinline__ unsigned packbf(float a, float b) {
    unsigned lo = (unsigned)__builtin_bit_cast(unsigned short, tobf(a));
    unsigned hi = (unsigned)__builtin_bit_cast(unsigned short, tobf(b));
    return lo | (hi << 16);
}

static __device__ __forceinline__ v8f wmma_bf16(v16bf a, v16bf b, v8f c) {
    return __builtin_amdgcn_wmma_f32_16x16x32_bf16(false, a, false, b, (short)0, c, false, false);
}

// A-fragment element pair base K for vgpr v (contiguous order k = 16*half + s, wave32)
static __device__ __forceinline__ int kpair(int v, int lane) {
    return ((lane >> 4) << 4) + (v << 1);
}

// A-fragment from bf16 row-major row pointer (row = M of this lane), K offset k0
static __device__ __forceinline__ v16bf load_afrag(const __bf16* row, int k0, int lane) {
    v16bf a;
#pragma unroll
    for (int v = 0; v < 8; ++v)
        ((unsigned*)&a)[v] = *(const unsigned*)(row + k0 + kpair(v, lane));
    return a;
}

// B-fragment: 16 contiguous bf16 (32B, 16B-aligned) at p
static __device__ __forceinline__ v16bf load_bfrag(const __bf16* p) {
    v16bf r;
    ((u32x4*)&r)[0] = *(const u32x4*)(p);
    ((u32x4*)&r)[1] = *(const u32x4*)(p + 8);
    return r;
}
static __device__ __forceinline__ v16bf zero_frag() {
    v16bf r;
#pragma unroll
    for (int v = 0; v < 8; ++v) ((unsigned*)&r)[v] = 0u;
    return r;
}

// ================= kernel 0: weight conversion + offsets =================
__global__ __launch_bounds__(256) void prep_kernel(const float* __restrict__ w_in,
                                                   const float* __restrict__ w_out,
                                                   const int* __restrict__ lengths,
                                                   __bf16* __restrict__ w16in,
                                                   __bf16* __restrict__ w16out,
                                                   int* __restrict__ offs) {
    int idx = blockIdx.x * 256 + threadIdx.x;
    if (idx < F_ * E_) w16in[idx] = tobf(w_in[idx]);
    int j = idx - F_ * E_;
    if (j >= 0 && j < E_ * E_) w16out[j] = tobf(w_out[j]);
    if (idx == 0) {
        int o = 0;
        for (int b = 0; b < B_; ++b) { offs[b] = o; o += lengths[b]; }
    }
}

// ================= kernel 1: layernorm -> bf16 =================
__global__ __launch_bounds__(256) void ln_kernel(const float* __restrict__ x,
                                                 const float* __restrict__ lw,
                                                 const float* __restrict__ lb,
                                                 __bf16* __restrict__ xn) {
    int row  = blockIdx.x * 8 + (threadIdx.x >> 5);
    int lane = threadIdx.x & 31;
    const float* xr = x + (size_t)row * E_ + lane * 16;
    float v[16], s = 0.f;
#pragma unroll
    for (int i = 0; i < 16; ++i) { v[i] = xr[i]; s += v[i]; }
#pragma unroll
    for (int m = 1; m < 32; m <<= 1) s += __shfl_xor(s, m, 32);
    float mu = s * (1.0f / E_);
    float q = 0.f;
#pragma unroll
    for (int i = 0; i < 16; ++i) { float d = v[i] - mu; q += d * d; }
#pragma unroll
    for (int m = 1; m < 32; m <<= 1) q += __shfl_xor(q, m, 32);
    float rstd = rsqrtf(q * (1.0f / E_) + 1e-5f);
    __bf16* orow = xn + (size_t)row * E_ + lane * 16;
#pragma unroll
    for (int i = 0; i < 16; ++i) {
        int col = lane * 16 + i;
        orow[i] = tobf((v[i] - mu) * rstd * lw[col] + lb[col]);
    }
}

// ================= kernel 2: QKV GEMM (WMMA), 32Mx64N per wave =================
// qkv[t,f] = sum_e xn[t,e]*w_in[f,e] + b_in[f]
__global__ __launch_bounds__(256) void qkv_kernel(const __bf16* __restrict__ xn,
                                                  const __bf16* __restrict__ w16,
                                                  const float* __restrict__ b_in,
                                                  __bf16* __restrict__ qkv) {
    int tile = blockIdx.x * 8 + (threadIdx.x >> 5);   // 12288 waves
    int lane = threadIdx.x & 31;
    int half = lane >> 4, lx = lane & 15;
    int mt = tile / 24, nt = tile % 24;               // 512 x 24
    const __bf16* arow0 = xn + (size_t)(mt * 32 + lx) * E_;
    const __bf16* arow1 = xn + (size_t)(mt * 32 + 16 + lx) * E_;
    const __bf16* brow0 = w16 + (size_t)(nt * 64 + lx) * E_ + half * 16;
    v8f acc[2][4];
#pragma unroll
    for (int i = 0; i < 2; ++i)
#pragma unroll
        for (int j = 0; j < 4; ++j)
#pragma unroll
            for (int r = 0; r < 8; ++r) acc[i][j][r] = 0.f;
#pragma unroll 4
    for (int k0 = 0; k0 < E_; k0 += 32) {
        v16bf a0 = load_afrag(arow0, k0, lane);
        v16bf a1 = load_afrag(arow1, k0, lane);
#pragma unroll
        for (int j = 0; j < 4; ++j) {
            v16bf b = load_bfrag(brow0 + (size_t)j * 16 * E_ + k0);
            acc[0][j] = wmma_bf16(a0, b, acc[0][j]);
            acc[1][j] = wmma_bf16(a1, b, acc[1][j]);
        }
    }
#pragma unroll
    for (int i = 0; i < 2; ++i)
#pragma unroll
        for (int j = 0; j < 4; ++j) {
            int col = nt * 64 + j * 16 + lx;
            float bias = b_in[col];
#pragma unroll
            for (int r = 0; r < 8; ++r) {
                int row = mt * 32 + i * 16 + r + half * 8;
                qkv[(size_t)row * F_ + col] = tobf(acc[i][j][r] + bias);
            }
        }
}

// ================= kernel 3: build padded V^T [B][H][D][LMAX] =================
__global__ __launch_bounds__(256) void vt_kernel(const __bf16* __restrict__ qkv,
                                                 const int* __restrict__ offs,
                                                 const int* __restrict__ lens,
                                                 __bf16* __restrict__ vt) {
    int idx = blockIdx.x * 256 + threadIdx.x;
    int m = idx % LMAX_;
    int r = idx / LMAX_;
    int d = r % D_; r /= D_;
    int h = r % H_;
    int b = r / H_;
    __bf16 z = __builtin_bit_cast(__bf16, (unsigned short)0);
    __bf16 val = z;
    if (m < lens[b])
        val = qkv[(size_t)(offs[b] + m) * F_ + 2 * E_ + h * D_ + d];
    vt[idx] = val;
}

// ================= kernel 4: attention (flash-style, WMMA + async V) =========
// block = (b, 16-query block); 8 waves = 8 heads
__global__ __launch_bounds__(256) void attn_kernel(const __bf16* __restrict__ qkv,
                                                   const __bf16* __restrict__ vt,
                                                   const float* __restrict__ b_in,
                                                   const int* __restrict__ offs,
                                                   const int* __restrict__ lens,
                                                   __bf16* __restrict__ out16,
                                                   float* __restrict__ attn_out) {
    __shared__ float Pc[8][16][33];          // per-wave P chunk (fp32), padded
#ifdef HAVE_ASYNC_LDS
    __shared__ __bf16 Vs[8][4][32][16];      // per-wave V chunk staging (32KB)
#endif
    int b  = blockIdx.x / 24;
    int lb = blockIdx.x % 24;
    int tid = threadIdx.x, wid = tid >> 5, lane = tid & 31;
    int h = wid;
    int half = lane >> 4, lx = lane & 15;
    int len = lens[b], off = offs[b];

    // ---- Q fragments (2 K-steps of 32), bias fallback for padded rows ----
    int qrow = lb * 16 + lx;
    bool qvalid = qrow < len;
    const __bf16* qptr = qkv + (size_t)(off + qrow) * F_ + h * D_;
    v16bf qf[2];
#pragma unroll
    for (int ks = 0; ks < 2; ++ks) {
#pragma unroll
        for (int v = 0; v < 8; ++v) {
            int k = ks * 32 + kpair(v, lane);
            unsigned p = qvalid ? *(const unsigned*)(qptr + k)
                                : packbf(b_in[h * D_ + k], b_in[h * D_ + k + 1]);
            ((unsigned*)&qf[ks])[v] = p;
        }
    }

    float runmax[8], rowsum[8];
    v8f oacc[4];
#pragma unroll
    for (int r = 0; r < 8; ++r) { runmax[r] = -1e30f; rowsum[r] = 0.f; }
#pragma unroll
    for (int dt = 0; dt < 4; ++dt)
#pragma unroll
        for (int i = 0; i < 8; ++i) oacc[dt][i] = 0.f;

    const __bf16* vbase = vt + ((size_t)(b * H_ + h)) * D_ * LMAX_;

    // ---- pass 1: online softmax + P*V ----
    for (int c = 0; c < 12; ++c) {
#ifdef HAVE_ASYNC_LDS
        // kick off async copy of this chunk's V tile into our LDS slab;
        // it overlaps with the Q*K^T WMMAs and the softmax VALU work below.
#pragma unroll
        for (int dt = 0; dt < 4; ++dt) {
            const __bf16* vp = vbase + (size_t)(dt * 16 + lx) * LMAX_ + c * 32 + half * 16;
            gv4i* g = (gv4i*)vp;
            lv4i* l = (lv4i*)&Vs[wid][dt][lane][0];
            __builtin_amdgcn_global_load_async_to_lds_b128(g, l, 0, 0);
            __builtin_amdgcn_global_load_async_to_lds_b128(g + 1, l + 1, 0, 0);
        }
#endif
        v8f s[2];
#pragma unroll
        for (int t = 0; t < 2; ++t) {
            int m = c * 32 + t * 16 + lx;
            bool kv = m < len;
            const __bf16* kptr = qkv + (size_t)(off + m) * F_ + E_ + h * D_ + half * 16;
            v8f acc;
#pragma unroll
            for (int i = 0; i < 8; ++i) acc[i] = 0.f;
#pragma unroll
            for (int ks = 0; ks < 2; ++ks) {
                v16bf bf = kv ? load_bfrag(kptr + ks * 32) : zero_frag();
                acc = wmma_bf16(qf[ks], bf, acc);
            }
            s[t] = acc;
        }
        int m0 = c * 32 + lx, m1 = c * 32 + 16 + lx;
#pragma unroll
        for (int r = 0; r < 8; ++r) {
            float s0 = (m0 < len) ? s[0][r] * 0.125f : -1e9f;
            float s1 = (m1 < len) ? s[1][r] * 0.125f : -1e9f;
            float cm = fmaxf(s0, s1);
#pragma unroll
            for (int mk = 1; mk < 16; mk <<= 1) cm = fmaxf(cm, __shfl_xor(cm, mk, 32));
            float nm = fmaxf(runmax[r], cm);
            float alpha = __expf(runmax[r] - nm);
            float p0 = __expf(s0 - nm), p1 = __expf(s1 - nm);
            float ps = p0 + p1;
#pragma unroll
            for (int mk = 1; mk < 16; mk <<= 1) ps += __shfl_xor(ps, mk, 32);
            rowsum[r] = rowsum[r] * alpha + ps;
            runmax[r] = nm;
#pragma unroll
            for (int dt = 0; dt < 4; ++dt) oacc[dt][r] *= alpha;
            Pc[wid][r + half * 8][lx]      = p0;
            Pc[wid][r + half * 8][16 + lx] = p1;
        }
        __syncthreads();
        v16bf pf;
#pragma unroll
        for (int v = 0; v < 8; ++v) {
            int k = kpair(v, lane);
            ((unsigned*)&pf)[v] = packbf(Pc[wid][lx][k], Pc[wid][lx][k + 1]);
        }
#ifdef HAVE_ASYNC_LDS
#if __has_builtin(__builtin_amdgcn_s_wait_asynccnt)
        __builtin_amdgcn_s_wait_asynccnt(0);
#else
        asm volatile("s_wait_asynccnt 0x0" ::: "memory");
#endif
#endif
#pragma unroll
        for (int dt = 0; dt < 4; ++dt) {
            v16bf bf;
#ifdef HAVE_ASYNC_LDS
            ((u32x4*)&bf)[0] = *(const u32x4*)&Vs[wid][dt][lane][0];
            ((u32x4*)&bf)[1] = *(const u32x4*)&Vs[wid][dt][lane][8];
#else
            bf = load_bfrag(vbase + (size_t)(dt * 16 + lx) * LMAX_ + c * 32 + half * 16);
#endif
            oacc[dt] = wmma_bf16(pf, bf, oacc[dt]);
        }
        __syncthreads();
    }

    // ---- write O (zero padded rows) ----
    __bf16 z = __builtin_bit_cast(__bf16, (unsigned short)0);
#pragma unroll
    for (int dt = 0; dt < 4; ++dt) {
#pragma unroll
        for (int r = 0; r < 8; ++r) {
            int l = lb * 16 + r + half * 8;
            float o = oacc[dt][r] / rowsum[r];
            out16[((size_t)b * LMAX_ + l) * E_ + h * D_ + dt * 16 + lx] =
                (l < len) ? tobf(o) : z;
        }
    }

    // ---- pass 2: emit normalized attention weights (mean over heads) ----
    for (int c = 0; c < 12; ++c) {
#pragma unroll
        for (int t = 0; t < 2; ++t) {
            int m = c * 32 + t * 16 + lx;
            bool kv = m < len;
            const __bf16* kptr = qkv + (size_t)(off + m) * F_ + E_ + h * D_ + half * 16;
            v8f acc;
#pragma unroll
            for (int i = 0; i < 8; ++i) acc[i] = 0.f;
#pragma unroll
            for (int ks = 0; ks < 2; ++ks) {
                v16bf bf = kv ? load_bfrag(kptr + ks * 32) : zero_frag();
                acc = wmma_bf16(qf[ks], bf, acc);
            }
#pragma unroll
            for (int r = 0; r < 8; ++r) {
                float sv = kv ? acc[r] * 0.125f : -1e9f;
                float p  = __expf(sv - runmax[r]) / rowsum[r];
                Pc[wid][r + half * 8][t * 16 + lx] = p;
            }
        }
        __syncthreads();
#pragma unroll
        for (int i = 0; i < 2; ++i) {
            int pos = tid + i * 256;
            int l = pos >> 5, ml = pos & 31;
            float sum = 0.f;
#pragma unroll
            for (int hh = 0; hh < 8; ++hh) sum += Pc[hh][l][ml];
            attn_out[((size_t)b * LMAX_ + lb * 16 + l) * LMAX_ + c * 32 + ml] = sum * 0.125f;
        }
        __syncthreads();
    }
}

// ================= kernel 5: out-proj + bias + residual, 32Mx64N per wave =====
__global__ __launch_bounds__(256) void outproj_kernel(const __bf16* __restrict__ out16,
                                                      const __bf16* __restrict__ w16o,
                                                      const float* __restrict__ b_out,
                                                      const float* __restrict__ x,
                                                      const int* __restrict__ offs,
                                                      const int* __restrict__ lens,
                                                      float* __restrict__ y) {
    int tile = blockIdx.x * 8 + (threadIdx.x >> 5);   // 6144 waves
    int lane = threadIdx.x & 31;
    int half = lane >> 4, lx = lane & 15;
    int nt = tile & 7;                                 // 8 col-strips of 64
    int mt = tile >> 3;                                // 768 row-strips of 32
    int b = mt / 24, lb2 = mt % 24;                    // lb2 = 32-row block
    const __bf16* arow0 = out16 + ((size_t)mt * 32 + lx) * E_;
    const __bf16* arow1 = out16 + ((size_t)mt * 32 + 16 + lx) * E_;
    const __bf16* brow0 = w16o + (size_t)(nt * 64 + lx) * E_ + half * 16;
    v8f acc[2][4];
#pragma unroll
    for (int i = 0; i < 2; ++i)
#pragma unroll
        for (int j = 0; j < 4; ++j)
#pragma unroll
            for (int r = 0; r < 8; ++r) acc[i][j][r] = 0.f;
#pragma unroll 4
    for (int k0 = 0; k0 < E_; k0 += 32) {
        v16bf a0 = load_afrag(arow0, k0, lane);
        v16bf a1 = load_afrag(arow1, k0, lane);
#pragma unroll
        for (int j = 0; j < 4; ++j) {
            v16bf bf = load_bfrag(brow0 + (size_t)j * 16 * E_ + k0);
            acc[0][j] = wmma_bf16(a0, bf, acc[0][j]);
            acc[1][j] = wmma_bf16(a1, bf, acc[1][j]);
        }
    }
    int len = lens[b], off = offs[b];
#pragma unroll
    for (int i = 0; i < 2; ++i)
#pragma unroll
        for (int j = 0; j < 4; ++j) {
            int col = nt * 64 + j * 16 + lx;
            float bias = b_out[col];
#pragma unroll
            for (int r = 0; r < 8; ++r) {
                int l = lb2 * 32 + i * 16 + r + half * 8;
                if (l < len) {
                    size_t t = (size_t)(off + l);
                    y[t * E_ + col] = acc[i][j][r] + bias + x[t * E_ + col];
                }
            }
        }
}

// ================= host launcher =================
extern "C" void kernel_launch(void* const* d_in, const int* in_sizes, int n_in,
                              void* d_out, int out_size, void* d_ws, size_t ws_size,
                              hipStream_t stream) {
    const float* x     = (const float*)d_in[0];
    const int*   lens  = (const int*)d_in[1];
    const float* ln_w  = (const float*)d_in[2];
    const float* ln_b  = (const float*)d_in[3];
    const float* w_in  = (const float*)d_in[4];
    const float* b_in  = (const float*)d_in[5];
    const float* w_out = (const float*)d_in[6];
    const float* b_out = (const float*)d_in[7];

    char* ws = (char*)d_ws;
    __bf16* xn16   = (__bf16*)(ws + WS_XN);
    __bf16* qkv16  = (__bf16*)(ws + WS_QKV);
    __bf16* vt     = (__bf16*)(ws + WS_VT);
    __bf16* out16  = (__bf16*)(ws + WS_OUT16);
    __bf16* w16in  = (__bf16*)(ws + WS_WIN);
    __bf16* w16out = (__bf16*)(ws + WS_WOUT);
    int*    offs   = (int*)(ws + WS_OFFS);

    float* y        = (float*)d_out;
    float* attn_out = (float*)d_out + (size_t)T_ * E_;

    hipLaunchKernelGGL(prep_kernel, dim3(4096), dim3(256), 0, stream,
                       w_in, w_out, lens, w16in, w16out, offs);
    hipLaunchKernelGGL(ln_kernel, dim3(T_ / 8), dim3(256), 0, stream,
                       x, ln_w, ln_b, xn16);
    hipLaunchKernelGGL(qkv_kernel, dim3((T_ / 32) * (F_ / 64) / 8), dim3(256), 0, stream,
                       xn16, w16in, b_in, qkv16);
    hipLaunchKernelGGL(vt_kernel, dim3(B_ * H_ * D_ * LMAX_ / 256), dim3(256), 0, stream,
                       qkv16, offs, lens, vt);
    hipLaunchKernelGGL(attn_kernel, dim3(B_ * (LMAX_ / 16)), dim3(256), 0, stream,
                       qkv16, vt, b_in, offs, lens, out16, attn_out);
    hipLaunchKernelGGL(outproj_kernel, dim3((B_ * LMAX_ / 32) * (E_ / 64) / 8), dim3(256), 0, stream,
                       out16, w16out, b_out, x, offs, lens, y);
    (void)in_sizes; (void)n_in; (void)out_size; (void)ws_size;
}
